// SparseMSDeformableAttention_8297876815876
// MI455X (gfx1250) — matrix operand order repaired and checked by reading, not compiled, not run on hardware
//
#include <hip/hip_runtime.h>
#include <math.h>

#define Q_TOTAL 8192
#define EMBED   256
#define NLVL    4
#define NPTS    4
#define NHEADS  8
#define HDIM    32
#define NOFF    256   // N_LEVELS*N_POINTS*N_HEADS*2
#define NATT    128   // N_LEVELS*N_POINTS*N_HEADS
#define NCAT    384   // NOFF + NATT
#define HMAXD   128
#define WMAXD   128
#define EPSV    1e-5f

typedef _Float16 h4   __attribute__((ext_vector_type(4)));
typedef _Float16 h8   __attribute__((ext_vector_type(8)));
typedef _Float16 v16h __attribute__((ext_vector_type(16)));
typedef float    f4   __attribute__((ext_vector_type(4)));
typedef float    v8f  __attribute__((ext_vector_type(8)));

union V16U { v16h v; h8 h[2]; };

__device__ inline int clampi(int v, int lo, int hi) {
    v = v < lo ? lo : v;
    return v > hi ? hi : v;
}

// ---------------------------------------------------------------------------
// Prep kernels: f32 -> f16 conversion, weight transpose to [N][K] row-major
// ---------------------------------------------------------------------------
__global__ void k_cvt_query(const float* __restrict__ q, _Float16* __restrict__ qh, int n) {
    int i = blockIdx.x * blockDim.x + threadIdx.x;
    if (i < n) qh[i] = (_Float16)q[i];
}

__global__ void k_prep_wcat(const float* __restrict__ Woff, const float* __restrict__ Watt,
                            _Float16* __restrict__ wt) {
    int i = blockIdx.x * blockDim.x + threadIdx.x;  // NCAT*EMBED threads
    int n = i >> 8, k = i & 255;
    float v = (n < NOFF) ? Woff[(size_t)k * NOFF + n] : Watt[(size_t)k * NATT + (n - NOFF)];
    wt[(size_t)n * EMBED + k] = (_Float16)v;
}

__global__ void k_prep_wout(const float* __restrict__ Wout, _Float16* __restrict__ wt) {
    int i = blockIdx.x * blockDim.x + threadIdx.x;  // EMBED*EMBED threads
    int n = i >> 8, k = i & 255;
    wt[(size_t)n * EMBED + k] = (_Float16)Wout[(size_t)k * EMBED + n];
}

// ---------------------------------------------------------------------------
// WMMA 16x16 tile over K=256 (8 steps of 16x16x32 f16 -> f32 accumulate).
// Per-lane fragment layouts per CDNA5 ISA 7.12.2:
//   A (16x32, M=lane&15): lanes 0-15 hold K = ks+[0..7] and ks+[16..23];
//                         lanes 16-31 hold K = ks+[8..15] and ks+[24..31].
//   B (32x16, N=lane&15): lanes 0-15 hold K = ks+[0..7] and ks+[16..23];
//                         lanes 16-31 hold K = ks+[8..15] and ks+[24..31].
// Both fragments are contiguous 16B chunks in [row][K] f16 memory.
// ---------------------------------------------------------------------------
__device__ inline v8f wmma_row_256k(const _Float16* __restrict__ arow,
                                    const _Float16* __restrict__ brow,
                                    int khalf /* lane>>4 */) {
    v8f acc = {};
    const int kA = khalf * 8;
    const int kB = khalf * 8;
#pragma unroll
    for (int ks = 0; ks < EMBED; ks += 32) {
        V16U a, b;
        a.h[0] = *(const h8*)(arow + ks + kA);
        a.h[1] = *(const h8*)(arow + ks + kA + 16);
        b.h[0] = *(const h8*)(brow + ks + kB);
        b.h[1] = *(const h8*)(brow + ks + kB + 16);
        acc = __builtin_amdgcn_wmma_f32_16x16x32_f16(false, a.v, false, b.v,
                                                     (short)0, acc, false, false);
    }
    return acc;
}

// GEMM 1: P[Q,384] = query_f16[Q,256] @ Wcat^T  (raw logits, bias added later)
__global__ void k_gemm_proj(const _Float16* __restrict__ qh, const _Float16* __restrict__ wt,
                            float* __restrict__ P) {
    int mtile = blockIdx.x / 3;
    int nb    = blockIdx.x % 3;
    int wave  = threadIdx.x >> 5;
    int lane  = threadIdx.x & 31;
    int lrow  = lane & 15;
    int khalf = lane >> 4;
    int mbase = mtile * 16;
    int nbase = (nb * 8 + wave) * 16;
    const _Float16* arow = qh + (size_t)(mbase + lrow) * EMBED;
    const _Float16* brow = wt + (size_t)(nbase + lrow) * EMBED;
    v8f acc = wmma_row_256k(arow, brow, khalf);
    int n = nbase + lrow;
#pragma unroll
    for (int j = 0; j < 8; ++j) {
        int m = mbase + j + 8 * khalf;   // C/D layout: VGPR j -> M = j + 8*(lane>>4)
        P[(size_t)m * NCAT + n] = acc[j];
    }
}

// GEMM 2: out[Q,256] = mid_f16[Q,256] @ Wout^T + b_out
__global__ void k_gemm_out(const _Float16* __restrict__ mh, const _Float16* __restrict__ wot,
                           const float* __restrict__ bout, float* __restrict__ out) {
    int mtile = blockIdx.x >> 1;
    int nb    = blockIdx.x & 1;
    int wave  = threadIdx.x >> 5;
    int lane  = threadIdx.x & 31;
    int lrow  = lane & 15;
    int khalf = lane >> 4;
    int mbase = mtile * 16;
    int nbase = (nb * 8 + wave) * 16;
    const _Float16* arow = mh  + (size_t)(mbase + lrow) * EMBED;
    const _Float16* brow = wot + (size_t)(nbase + lrow) * EMBED;
    v8f acc = wmma_row_256k(arow, brow, khalf);
    int n = nbase + lrow;
    float bias = bout[n];
#pragma unroll
    for (int j = 0; j < 8; ++j) {
        int m = mbase + j + 8 * khalf;
        out[(size_t)m * EMBED + n] = acc[j] + bias;
    }
}

// ---------------------------------------------------------------------------
// Fused sampling. Mapping: 4 channels/lane (float4), 8 lanes/head, 4 heads
// per wave, 2 waves per query, 4 queries per 256-thread block.
//  - Every corner fetch is one global_load_b128 per wave serving 4 heads'
//    corners (512B/instr) -> 4x fewer VMEM instructions than b32/lane.
//  - Per-head scalar math (softmax, sigmoid, bilinear weights, addresses)
//    executes for 4 heads per wave in a single instruction stream.
// Value tensor (134MB) is L2-resident; default RT temporal hint is correct.
// ---------------------------------------------------------------------------
__global__ void k_sample(const float* __restrict__ P,
                         const float* __restrict__ refxy,
                         const float* __restrict__ V,
                         const float* __restrict__ boff,
                         const float* __restrict__ batt,
                         const int*   __restrict__ boffsets,
                         const int*   __restrict__ sshapes,
                         _Float16* __restrict__ midh) {
    int wave = threadIdx.x >> 5;
    int lane = threadIdx.x & 31;
    int q  = blockIdx.x * 4 + (wave >> 1);          // 4 queries per block
    int h  = (wave & 1) * 4 + (lane >> 3);          // 4 heads per wave
    int d4 = (lane & 7) * 4;                        // 4 channels per lane
    const float* Pq = P + (size_t)q * NCAT;

    // softmax over the 16 (level,point) logits of this lane's head
    float lg[16];
#pragma unroll
    for (int lp = 0; lp < 16; ++lp)
        lg[lp] = Pq[NOFF + lp * NHEADS + h] + batt[lp * NHEADS + h];
    float mx = lg[0];
#pragma unroll
    for (int lp = 1; lp < 16; ++lp) mx = fmaxf(mx, lg[lp]);
    float sum = 0.f;
#pragma unroll
    for (int lp = 0; lp < 16; ++lp) { lg[lp] = __expf(lg[lp] - mx); sum += lg[lp]; }
    float isum = 1.f / sum;

    // inverse-sigmoid of clipped reference point
    float rx = fminf(fmaxf(refxy[(size_t)q * 2 + 0], 0.f), 1.f);
    float ry = fminf(fmaxf(refxy[(size_t)q * 2 + 1], 0.f), 1.f);
    float ivx = logf(fminf(fmaxf(rx, EPSV), 1.f) / fminf(fmaxf(1.f - rx, EPSV), 1.f));
    float ivy = logf(fminf(fmaxf(ry, EPSV), 1.f) / fminf(fmaxf(1.f - ry, EPSV), 1.f));

    // batch index: searchsorted(batch_offsets, q, 'right') - 1
    int b = 0;
    for (int i = 1; i < 2; ++i) if (q >= boffsets[i]) b = i;
    const float* Vb = V + (size_t)b * HMAXD * WMAXD * NLVL * NHEADS * HDIM;

    f4 acc = {0.f, 0.f, 0.f, 0.f};
#pragma unroll
    for (int l = 0; l < NLVL; ++l) {
        float hsz = (float)sshapes[l * 2 + 0];
        float wsz = (float)sshapes[l * 2 + 1];
        int hcap = sshapes[l * 2 + 0] - 1;
        int wcap = sshapes[l * 2 + 1] - 1;
#pragma unroll
        for (int p = 0; p < NPTS; ++p) {
            int lp = l * NPTS + p;
            int nb2 = (lp * NHEADS + h) * 2;
            float ox = Pq[nb2 + 0] + boff[nb2 + 0];
            float oy = Pq[nb2 + 1] + boff[nb2 + 1];
            float locx = 2.f / (1.f + __expf(-(ivx + ox))) - 1.f;
            float locy = 2.f / (1.f + __expf(-(ivy + oy))) - 1.f;
            float x = ((locx + 1.f) * wsz - 1.f) * 0.5f;
            float y = ((locy + 1.f) * hsz - 1.f) * 0.5f;
            int fx = (int)floorf(x), fy = (int)floorf(y);
            // reference clips x-indices to width-1 and uses x as the first
            // spatial index of V — replicated exactly.
            int ix0 = clampi(fx,     0, wcap);
            int ix1 = clampi(fx + 1, 0, wcap);
            int iy0 = clampi(fy,     0, hcap);
            int iy1 = clampi(fy + 1, 0, hcap);
            float wa = ((float)ix1 - x) * ((float)iy1 - y);
            float wb = ((float)ix1 - x) * (y - (float)iy0);
            float wc = (x - (float)ix0) * ((float)iy1 - y);
            float wd = (x - (float)ix0) * (y - (float)iy0);
            float aww = lg[lp] * isum;
            size_t cbase = (size_t)(l * NHEADS + h) * HDIM + d4;      // 16B aligned
            const size_t stride = (size_t)NLVL * NHEADS * HDIM;      // per (x,y) cell
            f4 va = *(const f4*)(Vb + (size_t)(ix0 * WMAXD + iy0) * stride + cbase);
            f4 vb = *(const f4*)(Vb + (size_t)(ix0 * WMAXD + iy1) * stride + cbase);
            f4 vc = *(const f4*)(Vb + (size_t)(ix1 * WMAXD + iy0) * stride + cbase);
            f4 vd = *(const f4*)(Vb + (size_t)(ix1 * WMAXD + iy1) * stride + cbase);
            acc += (aww * wa) * va + (aww * wb) * vb + (aww * wc) * vc + (aww * wd) * vd;
        }
    }
    h4 o;
    o.x = (_Float16)acc.x; o.y = (_Float16)acc.y;
    o.z = (_Float16)acc.z; o.w = (_Float16)acc.w;
    *(h4*)(midh + (size_t)q * EMBED + h * HDIM + d4) = o;   // 8B store
}

// ---------------------------------------------------------------------------
extern "C" void kernel_launch(void* const* d_in, const int* in_sizes, int n_in,
                              void* d_out, int out_size, void* d_ws, size_t ws_size,
                              hipStream_t stream) {
    (void)in_sizes; (void)n_in; (void)out_size; (void)ws_size;
    const float* query   = (const float*)d_in[0];
    const float* refxy   = (const float*)d_in[1];
    const float* V       = (const float*)d_in[2];
    const float* Woff    = (const float*)d_in[3];
    const float* boff    = (const float*)d_in[4];
    const float* Watt    = (const float*)d_in[5];
    const float* batt    = (const float*)d_in[6];
    const float* Wout    = (const float*)d_in[7];
    const float* bout    = (const float*)d_in[8];
    const int*   boffs   = (const int*)d_in[9];
    const int*   sshapes = (const int*)d_in[10];
    float* out = (float*)d_out;

    char* ws = (char*)d_ws;
    _Float16* qh   = (_Float16*)ws;  ws += (size_t)Q_TOTAL * EMBED * sizeof(_Float16);
    _Float16* wt   = (_Float16*)ws;  ws += (size_t)NCAT * EMBED * sizeof(_Float16);
    _Float16* wot  = (_Float16*)ws;  ws += (size_t)EMBED * EMBED * sizeof(_Float16);
    float*    P    = (float*)ws;     ws += (size_t)Q_TOTAL * NCAT * sizeof(float);
    _Float16* midh = (_Float16*)ws;  ws += (size_t)Q_TOTAL * EMBED * sizeof(_Float16);

    k_cvt_query<<<(Q_TOTAL * EMBED) / 256, 256, 0, stream>>>(query, qh, Q_TOTAL * EMBED);
    k_prep_wcat<<<(NCAT * EMBED) / 256, 256, 0, stream>>>(Woff, Watt, wt);
    k_prep_wout<<<(EMBED * EMBED) / 256, 256, 0, stream>>>(Wout, wot);

    // GEMM 1: 512 M-tiles x 3 N-blocks (8 waves x 16 cols each = 128 cols/block)
    k_gemm_proj<<<(Q_TOTAL / 16) * 3, 256, 0, stream>>>(qh, wt, P);

    // Fused softmax + bilinear sampling: 4 queries per block, b128 gathers
    k_sample<<<Q_TOTAL / 4, 256, 0, stream>>>(P, refxy, V, boff, batt, boffs, sshapes, midh);

    // GEMM 2: 512 M-tiles x 2 N-blocks, bias fused in epilogue
    k_gemm_out<<<(Q_TOTAL / 16) * 2, 256, 0, stream>>>(midh, wot, bout, out);
}
